// DiffusionTransformerBlock_40175124087083
// MI455X (gfx1250) — matrix operand order repaired and checked by reading, not compiled, not run on hardware
//
#include <hip/hip_runtime.h>

typedef __bf16 bf16;
typedef __attribute__((ext_vector_type(16))) __bf16 v16bf;
typedef __attribute__((ext_vector_type(8)))  float  v8f;

#if __has_builtin(__builtin_amdgcn_global_load_async_to_lds_b128)
#define ASYNC_LDS 1
#else
#define ASYNC_LDS 0
#endif

// 16-byte int vector type matching the async-LDS builtin's parameter type
typedef __attribute__((__vector_size__(16))) int vi4;
typedef __attribute__((address_space(1))) vi4 vi4_g;   // global
typedef __attribute__((address_space(3))) vi4 vi4_l;   // LDS

__device__ __forceinline__ v8f zero_v8f() { v8f z = {}; return z; }

__device__ __forceinline__ unsigned short f2bfu(float f) {
    union { float f; unsigned u; } v; v.f = f;
    unsigned u = v.u + 0x7FFFu + ((v.u >> 16) & 1u);   // RNE
    return (unsigned short)(u >> 16);
}
__device__ __forceinline__ bf16 f2bf(float f) {
    union { unsigned short s; bf16 b; } o; o.s = f2bfu(f);
    return o.b;
}

// copy 16 bytes global -> LDS (async DMA on gfx1250)
__device__ __forceinline__ void cp16_g2l(bf16* l, const bf16* g) {
#if ASYNC_LDS
    vi4_g* gp = (vi4_g*)(unsigned long long)(const void*)g;
    vi4_l* lp = (vi4_l*)(unsigned)(unsigned long long)(void*)l;
    __builtin_amdgcn_global_load_async_to_lds_b128(gp, lp, 0, 0);
#else
    *reinterpret_cast<uint4*>(l) = *reinterpret_cast<const uint4*>(g);
#endif
}

template <int N>
__device__ __forceinline__ void async_wait_le() {
#if ASYNC_LDS
#if __has_builtin(__builtin_amdgcn_s_wait_asynccnt)
    __builtin_amdgcn_s_wait_asynccnt(N);
#else
    asm volatile("s_wait_asynccnt %0" :: "i"(N) : "memory");
#endif
#endif
}

__device__ __forceinline__ v8f wmma_bf16(v16bf a, v16bf b, v8f c) {
    return __builtin_amdgcn_wmma_f32_16x16x32_bf16(
        false, a, false, b, (short)0, c, false, false);
}

// A fragment (16x32, MxK): row = lane&15; VGPR i: k = (i>>2)*16 + (lane>>4)*8 + 2*(i&3)
__device__ __forceinline__ v16bf frag_a_lds(const bf16* base, int stride, int row_base, int kbase) {
    int lane = threadIdx.x & 31;
    int half = lane >> 4, r = lane & 15;
    union { v16bf vec; unsigned u[8]; } f;
#pragma unroll
    for (int i = 0; i < 8; ++i) {
        int k = kbase + ((i >> 2) << 4) + (half << 3) + ((i & 3) << 1);
        f.u[i] = *reinterpret_cast<const unsigned*>(base + (row_base + r) * stride + k);
    }
    return f.vec;
}

// B fragment (32x16, KxN): col n = nbase + (lane&15); VGPR i: k = (i>>2)*16 + (lane>>4)*8 + 2*(i&3)
__device__ __forceinline__ v16bf frag_b_lds(const bf16* base, int stride, int krow_base, int nbase) {
    int lane = threadIdx.x & 31;
    int half = lane >> 4, n = nbase + (lane & 15);
    union { v16bf vec; unsigned u[8]; } f;
#pragma unroll
    for (int i = 0; i < 8; ++i) {
        int k = krow_base + ((i >> 2) << 4) + (half << 3) + ((i & 3) << 1);
        f.vec[2 * i]     = base[k * stride + n];
        f.vec[2 * i + 1] = base[(k + 1) * stride + n];
    }
    return f.vec;
}

// ---------------------------------------------------------------- utilities

__global__ __launch_bounds__(256) void f2bf_kernel(const float4* __restrict__ in,
                                                   uint2* __restrict__ out, int n4) {
    for (int i = blockIdx.x * 256 + threadIdx.x; i < n4; i += gridDim.x * 256) {
        float4 v = in[i];
        uint2 o;
        o.x = (unsigned)f2bfu(v.x) | ((unsigned)f2bfu(v.y) << 16);
        o.y = (unsigned)f2bfu(v.z) | ((unsigned)f2bfu(v.w) << 16);
        out[i] = o;
    }
}

__global__ __launch_bounds__(256) void time_mlp_kernel(const float* __restrict__ temb,
                                                       const float* __restrict__ Wt,
                                                       const float* __restrict__ bt,
                                                       float* __restrict__ tp) {
    __shared__ float s[1024];
    int b = blockIdx.y;
    int c = blockIdx.x * 256 + threadIdx.x;
    for (int i = threadIdx.x; i < 1024; i += 256) {
        float t = temb[b * 1024 + i];
        s[i] = t / (1.0f + __expf(-t));
    }
    __syncthreads();
    float acc = bt[c];
    for (int k = 0; k < 1024; ++k)
        acc += s[k] * Wt[k * 6144 + c];
    tp[b * 6144 + c] = acc;
}

__global__ __launch_bounds__(256) void ln_mod_kernel(const float* __restrict__ x,
                                                     const float* __restrict__ g,
                                                     const float* __restrict__ be,
                                                     const float* __restrict__ tp,
                                                     int scale_off, int shift_off,
                                                     bf16* __restrict__ out) {
    __shared__ float red[256];
    int row = blockIdx.x;
    int b = row >> 11;
    int tid = threadIdx.x;
    const float* xr = x + (size_t)row * 1024;
    float4 xv = reinterpret_cast<const float4*>(xr)[tid];

    red[tid] = xv.x + xv.y + xv.z + xv.w;
    __syncthreads();
    for (int o = 128; o > 0; o >>= 1) {
        if (tid < o) red[tid] += red[tid + o];
        __syncthreads();
    }
    float mean = red[0] * (1.0f / 1024.0f);
    __syncthreads();

    float d0 = xv.x - mean, d1 = xv.y - mean, d2 = xv.z - mean, d3 = xv.w - mean;
    red[tid] = d0 * d0 + d1 * d1 + d2 * d2 + d3 * d3;
    __syncthreads();
    for (int o = 128; o > 0; o >>= 1) {
        if (tid < o) red[tid] += red[tid + o];
        __syncthreads();
    }
    float rstd = rsqrtf(red[0] * (1.0f / 1024.0f) + 1e-5f);

    const float* tpb = tp + b * 6144;
    float dv[4] = {d0, d1, d2, d3};
#pragma unroll
    for (int i = 0; i < 4; ++i) {
        int c = tid * 4 + i;
        float xn = dv[i] * rstd * g[c] + be[c];
        float val = xn * (1.0f + tpb[scale_off + c]) + tpb[shift_off + c];
        out[(size_t)row * 1024 + c] = f2bf(val);
    }
}

// ---------------------------------------------------------------- WMMA GEMM
// 128x128 block tile, K-step 32, 8 waves (4M x 2N), each wave 32x64 = 8 WMMAs/step
// Double-buffered LDS: async DMA for tile t+1 overlaps WMMAs of tile t.

#define SA_S 40    // 32 + 8 pad (bf16)
#define SB_S 136   // 128 + 8 pad

__global__ __launch_bounds__(256) void gemm_bf16_kernel(
    const bf16* __restrict__ A, const bf16* __restrict__ Bw,
    const float* __restrict__ bias,
    int M, int N, int K, int mode,
    float* __restrict__ out_f,
    bf16*  __restrict__ out_b,
    bf16*  __restrict__ qb, bf16* __restrict__ kb, bf16* __restrict__ vb,
    const float* __restrict__ resid,
    const float* __restrict__ tp) {

    __shared__ bf16 sA[2][128 * SA_S];
    __shared__ bf16 sB[2][32 * SB_S];

    int tid = threadIdx.x;
    int wave = tid >> 5, lane = tid & 31;
    int wm = wave & 3, wn = wave >> 2;
    int half = lane >> 4, nl = lane & 15;
    int m0 = blockIdx.y * 128, n0 = blockIdx.x * 128;

    const bf16* gA[2]; const bf16* gB[2];
    int aoff[2], boff[2];
#pragma unroll
    for (int i = 0; i < 2; ++i) {
        int idx = tid + i * 256;
        int r = idx >> 2, sg = idx & 3;            // A: 128 rows x 4 segs
        gA[i] = A + (size_t)(m0 + r) * K + sg * 8;
        aoff[i] = r * SA_S + sg * 8;
        int rb = idx >> 4, sb = idx & 15;          // B: 32 rows x 16 segs
        gB[i] = Bw + (size_t)rb * N + n0 + sb * 8;
        boff[i] = rb * SB_S + sb * 8;
    }

    v8f acc[2][4];
#pragma unroll
    for (int i = 0; i < 2; ++i)
#pragma unroll
        for (int j = 0; j < 4; ++j) acc[i][j] = zero_v8f();

    // prologue: stage tile 0 into buffer 0
#pragma unroll
    for (int i = 0; i < 2; ++i) {
        cp16_g2l(&sA[0][aoff[i]], gA[i]);
        cp16_g2l(&sB[0][boff[i]], gB[i]);
        gB[i] += (size_t)32 * N;
    }

    int nk = K >> 5;
    for (int t = 0; t < nk; ++t) {
        int cur = t & 1, nxt = cur ^ 1;
        if (t + 1 < nk) {
#pragma unroll
            for (int i = 0; i < 2; ++i) {
                cp16_g2l(&sA[nxt][aoff[i]], gA[i] + (t + 1) * 32);
                cp16_g2l(&sB[nxt][boff[i]], gB[i]);
                gB[i] += (size_t)32 * N;
            }
            async_wait_le<4>();   // current tile's 4 copies complete; next 4 in flight
        } else {
            async_wait_le<0>();
        }
        __syncthreads();

        v16bf a0 = frag_a_lds(sA[cur], SA_S, wm * 32, 0);
        v16bf a1 = frag_a_lds(sA[cur], SA_S, wm * 32 + 16, 0);
#pragma unroll
        for (int j = 0; j < 4; ++j) {
            v16bf bf = frag_b_lds(sB[cur], SB_S, 0, wn * 64 + j * 16);
            acc[0][j] = wmma_bf16(a0, bf, acc[0][j]);
            acc[1][j] = wmma_bf16(a1, bf, acc[1][j]);
        }
        __syncthreads();   // all reads of 'cur' done before t+1 overwrites it
    }

#pragma unroll
    for (int mi = 0; mi < 2; ++mi)
#pragma unroll
    for (int ni = 0; ni < 4; ++ni)
#pragma unroll
    for (int v = 0; v < 8; ++v) {
        int m = m0 + wm * 32 + mi * 16 + v + half * 8;
        int n = n0 + wn * 64 + ni * 16 + nl;
        float val = acc[mi][ni][v] + bias[n];
        if (mode == 0) {
            int part = n >> 10, rem = n & 1023;
            int h = rem >> 6, hd = rem & 63;
            int b = m >> 11, l = m & 2047;
            int bh = b * 16 + h;
            bf16 bv = f2bf(val);
            if (part == 0)       qb[((size_t)bh * 2048 + l) * 64 + hd] = bv;
            else if (part == 1)  kb[((size_t)bh * 64 + hd) * 2048 + l] = bv;
            else                 vb[((size_t)bh * 2048 + l) * 64 + hd] = bv;
        } else if (mode == 1) {
            int b = m >> 11;
            out_f[(size_t)m * 1024 + n] = resid[(size_t)m * 1024 + n]
                                        + tp[b * 6144 + 2048 + n] * val;
        } else if (mode == 2) {
            float gl = 0.5f * val * (1.0f + erff(val * 0.70710678118654752f));
            out_b[(size_t)m * N + n] = f2bf(gl);
        } else {
            int b = m >> 11;
            out_f[(size_t)m * 1024 + n] = resid[(size_t)m * 1024 + n]
                                        + tp[b * 6144 + 5120 + n] * val;
        }
    }
}

// ---------------------------------------------------------------- attention
// Double-buffered KT/V tiles: next key-block DMA overlaps S/softmax/PV phase.

#define LT_S 72    // 64 + 8 pad (bf16)
#define SS_S 66    // f32 score stride

__global__ __launch_bounds__(256) void attn_kernel(const bf16* __restrict__ qb,
                                                   const bf16* __restrict__ kb,
                                                   const bf16* __restrict__ vb,
                                                   bf16* __restrict__ ob) {
    __shared__ bf16  sQ[64 * LT_S];
    __shared__ bf16  sKT[2][64 * LT_S];
    __shared__ bf16  sV[2][64 * LT_S];
    __shared__ bf16  sP[64 * LT_S];
    __shared__ float sS[64 * SS_S];
    __shared__ float m_i[64], l_i[64], alpha_s[64];
    __shared__ float pmax[64 * 4], psum[64 * 4];

    int tid = threadIdx.x;
    int wave = tid >> 5, lane = tid & 31;
    int wm = wave & 3, wn = wave >> 2;
    int half = lane >> 4, nl = lane & 15;
    int bh = blockIdx.y;
    int m0 = blockIdx.x * 64;

    int rr[2], sg[2], toff[2];
    const bf16* gKT[2]; const bf16* gV[2];
#pragma unroll
    for (int i = 0; i < 2; ++i) {
        int idx = tid + i * 256;
        rr[i] = idx >> 3; sg[i] = idx & 7;
        toff[i] = rr[i] * LT_S + sg[i] * 8;
        gKT[i] = kb + ((size_t)bh * 64 + rr[i]) * 2048 + sg[i] * 8;   // + j0
        gV[i]  = vb + ((size_t)bh * 2048 + rr[i]) * 64 + sg[i] * 8;   // + j0*64
    }

    // prologue: Q tile + key-block 0 into buffer 0
#pragma unroll
    for (int i = 0; i < 2; ++i) {
        cp16_g2l(sQ + toff[i], qb + ((size_t)bh * 2048 + m0 + rr[i]) * 64 + sg[i] * 8);
        cp16_g2l(&sKT[0][toff[i]], gKT[i]);
        cp16_g2l(&sV[0][toff[i]],  gV[i]);
    }
    if (tid < 64) { m_i[tid] = -1e30f; l_i[tid] = 0.0f; }
    async_wait_le<0>();
    __syncthreads();

    v16bf qf0 = frag_a_lds(sQ, LT_S, wm * 16, 0);
    v16bf qf1 = frag_a_lds(sQ, LT_S, wm * 16, 32);

    v8f o0 = zero_v8f(), o1 = zero_v8f();

    for (int t = 0; t < 32; ++t) {
        int cur = t & 1, nxt = cur ^ 1;
        __syncthreads();   // all waves done reading buffer 'nxt' (used at t-1)
        if (t + 1 < 32) {
            int j1 = (t + 1) * 64;
#pragma unroll
            for (int i = 0; i < 2; ++i) {
                cp16_g2l(&sKT[nxt][toff[i]], gKT[i] + j1);
                cp16_g2l(&sV[nxt][toff[i]],  gV[i] + (size_t)j1 * 64);
            }
            async_wait_le<4>();   // 'cur' copies complete; 'nxt' in flight
        } else {
            async_wait_le<0>();
        }
        __syncthreads();

        v8f s0 = zero_v8f(), s1 = zero_v8f();
        s0 = wmma_bf16(qf0, frag_b_lds(sKT[cur], LT_S, 0,  wn * 32),      s0);
        s0 = wmma_bf16(qf1, frag_b_lds(sKT[cur], LT_S, 32, wn * 32),      s0);
        s1 = wmma_bf16(qf0, frag_b_lds(sKT[cur], LT_S, 0,  wn * 32 + 16), s1);
        s1 = wmma_bf16(qf1, frag_b_lds(sKT[cur], LT_S, 32, wn * 32 + 16), s1);
#pragma unroll
        for (int v = 0; v < 8; ++v) {
            int mr = wm * 16 + v + half * 8;
            sS[mr * SS_S + wn * 32 + nl]      = s0[v] * 0.125f;
            sS[mr * SS_S + wn * 32 + 16 + nl] = s1[v] * 0.125f;
        }
        __syncthreads();

        {
            int r = tid >> 2, q = tid & 3;
            float mx = -1e30f;
#pragma unroll
            for (int n = 0; n < 16; ++n)
                mx = fmaxf(mx, sS[r * SS_S + q * 16 + n]);
            pmax[r * 4 + q] = mx;
        }
        __syncthreads();
        if (tid < 64) {
            float mx = fmaxf(fmaxf(pmax[tid * 4], pmax[tid * 4 + 1]),
                             fmaxf(pmax[tid * 4 + 2], pmax[tid * 4 + 3]));
            float mnew = fmaxf(m_i[tid], mx);
            float a = __expf(m_i[tid] - mnew);
            alpha_s[tid] = a;
            m_i[tid] = mnew;
            l_i[tid] *= a;
        }
        __syncthreads();
        {
            int r = tid >> 2, q = tid & 3;
            float mnew = m_i[r];
            float sum = 0.0f;
#pragma unroll
            for (int n = 0; n < 16; ++n) {
                float p = __expf(sS[r * SS_S + q * 16 + n] - mnew);
                sP[r * LT_S + q * 16 + n] = f2bf(p);
                sum += p;
            }
            psum[r * 4 + q] = sum;
        }
        __syncthreads();
        if (tid < 64)
            l_i[tid] += psum[tid * 4] + psum[tid * 4 + 1] + psum[tid * 4 + 2] + psum[tid * 4 + 3];

#pragma unroll
        for (int v = 0; v < 8; ++v) {
            float a = alpha_s[wm * 16 + v + half * 8];
            o0[v] *= a; o1[v] *= a;
        }
        v16bf pf0 = frag_a_lds(sP, LT_S, wm * 16, 0);
        v16bf pf1 = frag_a_lds(sP, LT_S, wm * 16, 32);
        o0 = wmma_bf16(pf0, frag_b_lds(sV[cur], LT_S, 0,  wn * 32),      o0);
        o0 = wmma_bf16(pf1, frag_b_lds(sV[cur], LT_S, 32, wn * 32),      o0);
        o1 = wmma_bf16(pf0, frag_b_lds(sV[cur], LT_S, 0,  wn * 32 + 16), o1);
        o1 = wmma_bf16(pf1, frag_b_lds(sV[cur], LT_S, 32, wn * 32 + 16), o1);
    }
    __syncthreads();

    int b = bh >> 4, h = bh & 15;
#pragma unroll
    for (int v = 0; v < 8; ++v) {
        int rl = wm * 16 + v + half * 8;
        float inv = 1.0f / l_i[rl];
        size_t row = (size_t)b * 2048 + m0 + rl;
        ob[row * 1024 + h * 64 + wn * 32 + nl]      = f2bf(o0[v] * inv);
        ob[row * 1024 + h * 64 + wn * 32 + 16 + nl] = f2bf(o1[v] * inv);
    }
}

// ---------------------------------------------------------------- launch

extern "C" void kernel_launch(void* const* d_in, const int* in_sizes, int n_in,
                              void* d_out, int out_size, void* d_ws, size_t ws_size,
                              hipStream_t stream) {
    const float* x     = (const float*)d_in[0];
    const float* temb  = (const float*)d_in[1];
    const float* Wqkv  = (const float*)d_in[2];
    const float* bqkv  = (const float*)d_in[3];
    const float* Wproj = (const float*)d_in[4];
    const float* bproj = (const float*)d_in[5];
    const float* W1    = (const float*)d_in[6];
    const float* b1    = (const float*)d_in[7];
    const float* W2    = (const float*)d_in[8];
    const float* b2    = (const float*)d_in[9];
    const float* Wt    = (const float*)d_in[10];
    const float* bt    = (const float*)d_in[11];
    const float* g1    = (const float*)d_in[12];
    const float* be1   = (const float*)d_in[13];
    const float* g2    = (const float*)d_in[14];
    const float* be2   = (const float*)d_in[15];

    char* ws = (char*)d_ws;
    size_t o = 0;
    float* tp = (float*)(ws + o);          o += 4 * 6144 * 4;
    bf16* wqkv_b  = (bf16*)(ws + o);       o += (size_t)1024 * 3072 * 2;
    bf16* wproj_b = (bf16*)(ws + o);       o += (size_t)1024 * 1024 * 2;
    bf16* w1_b    = (bf16*)(ws + o);       o += (size_t)1024 * 4096 * 2;
    bf16* w2_b    = (bf16*)(ws + o);       o += (size_t)4096 * 1024 * 2;
    size_t o_region = o;
    bf16* h1    = (bf16*)(ws + o_region);
    bf16* qbuf  = (bf16*)(ws + o_region + (size_t)16777216);
    bf16* ktbuf = (bf16*)(ws + o_region + (size_t)2 * 16777216);
    bf16* vbuf  = (bf16*)(ws + o_region + (size_t)3 * 16777216);
    bf16* mid   = (bf16*)(ws + o_region);          // aliases h1/q/kt/v (dead by then)
    o = o_region + (size_t)4 * 16777216;
    bf16* attn = (bf16*)(ws + o);          o += (size_t)8192 * 1024 * 2;
    float* x1  = (float*)(ws + o);         o += (size_t)8192 * 1024 * 4;
    bf16* h2   = (bf16*)(ws + o);          o += (size_t)8192 * 1024 * 2;

    f2bf_kernel<<<2048, 256, 0, stream>>>((const float4*)Wqkv,  (uint2*)wqkv_b,  1024 * 3072 / 4);
    f2bf_kernel<<<2048, 256, 0, stream>>>((const float4*)Wproj, (uint2*)wproj_b, 1024 * 1024 / 4);
    f2bf_kernel<<<2048, 256, 0, stream>>>((const float4*)W1,    (uint2*)w1_b,    1024 * 4096 / 4);
    f2bf_kernel<<<2048, 256, 0, stream>>>((const float4*)W2,    (uint2*)w2_b,    4096 * 1024 / 4);

    time_mlp_kernel<<<dim3(24, 4), 256, 0, stream>>>(temb, Wt, bt, tp);
    ln_mod_kernel<<<8192, 256, 0, stream>>>(x, g1, be1, tp, 1024, 0, h1);

    gemm_bf16_kernel<<<dim3(24, 64), 256, 0, stream>>>(
        h1, wqkv_b, bqkv, 8192, 3072, 1024, 0,
        nullptr, nullptr, qbuf, ktbuf, vbuf, nullptr, nullptr);

    attn_kernel<<<dim3(32, 64), 256, 0, stream>>>(qbuf, ktbuf, vbuf, attn);

    gemm_bf16_kernel<<<dim3(8, 64), 256, 0, stream>>>(
        attn, wproj_b, bproj, 8192, 1024, 1024, 1,
        x1, nullptr, nullptr, nullptr, nullptr, x, tp);

    ln_mod_kernel<<<8192, 256, 0, stream>>>(x1, g2, be2, tp, 4096, 3072, h2);

    gemm_bf16_kernel<<<dim3(32, 64), 256, 0, stream>>>(
        h2, w1_b, b1, 8192, 4096, 1024, 2,
        nullptr, mid, nullptr, nullptr, nullptr, nullptr, nullptr);

    gemm_bf16_kernel<<<dim3(8, 64), 256, 0, stream>>>(
        mid, w2_b, b2, 8192, 1024, 4096, 3,
        (float*)d_out, nullptr, nullptr, nullptr, nullptr, x1, tp);

    (void)in_sizes; (void)n_in; (void)out_size; (void)ws_size;
}
